// MCNET_28991029248224
// MI455X (gfx1250) — 8-bit matrix operand order repaired and checked by reading, compiled, not run on hardware
//
#include <hip/hip_runtime.h>
#include <stdint.h>

typedef int v8i __attribute__((ext_vector_type(8)));

__device__ __forceinline__ signed char fsign8(float v) {
    return (signed char)((v > 0.0f) ? 1 : ((v < 0.0f) ? -1 : 0));
}

__device__ __forceinline__ signed char clamp8(int q) {
    return (signed char)(q > 1 ? 1 : (q < -1 ? -1 : q));
}

__global__ __launch_bounds__(256) void bpack2_kernel(const float* __restrict__ w2,
                                                     unsigned int* __restrict__ bp) {
    int idx = blockIdx.x * 256 + threadIdx.x;
    if (idx >= 1536) return;
    int v    = idx & 7;
    int lane = (idx >> 3) & 31;
    int st   = idx >> 8;
    int s = st % 3, t = st / 3;
    int oc = t * 16 + (lane & 15);
    int h  = lane >> 4;
    int k0 = s * 64 + (v >> 1) * 16 + h * 8 + (v & 1) * 4;
    unsigned int word = 0u;
    for (int j = 0; j < 4; ++j) {
        int k = k0 + j;
        signed char sb = 0;
        if (k < 144 && oc < 23) {
            int p = k >> 4, ic = k & 15;
            int ky = p / 3, kx = p % 3;
            sb = fsign8(w2[((oc * 16 + ic) * 3 + ky) * 3 + kx]);
        }
        word |= ((unsigned int)(unsigned char)sb) << (8 * j);
    }
    bp[idx] = word;
}

__global__ __launch_bounds__(256) void bpack3_kernel(const float* __restrict__ w3,
                                                     unsigned int* __restrict__ bp) {
    int idx = blockIdx.x * 256 + threadIdx.x;
    if (idx >= 1024) return;
    int v    = idx & 7;
    int lane = (idx >> 3) & 31;
    int s    = idx >> 8;
    int oc = lane & 15;
    int h  = lane >> 4;
    int k0 = s * 64 + (v >> 1) * 16 + h * 8 + (v & 1) * 4;
    unsigned int word = 0u;
    for (int j = 0; j < 4; ++j) {
        int k = k0 + j;
        int d = k >> 2;
        int q = ((d >> 2) << 1) + (d & 1);
        signed char sb = 0;
        if (q <= 26 && oc < 2) {
            int p = q / 3;
            int ky = p / 3, kx = p % 3;
            int ic = (q % 3) * 8 + ((d >> 1) & 1) * 4 + (k & 3);
            if (ic < 23) sb = fsign8(w3[((oc * 23 + ic) * 3 + ky) * 3 + kx]);
        }
        word |= ((unsigned int)(unsigned char)sb) << (8 * j);
    }
    bp[idx] = word;
}

__global__ __launch_bounds__(256) void layer1_kernel(const float* __restrict__ in,
                                                     const float* __restrict__ w1,
                                                     signed char* __restrict__ s1) {
    __shared__ float ws[432];
    int tid = threadIdx.x;
    for (int i = tid; i < 432; i += 256) {
        float f = w1[i];
        ws[i] = (f > 0.0f) ? 1.0f : ((f < 0.0f) ? -1.0f : 0.0f);
    }
    __syncthreads();
    int pid = blockIdx.x * 256 + tid;
    const int total = 32 * 510 * 510;
    if (pid >= total) return;
    int x = pid % 510;
    int y = (pid / 510) % 510;
    int n = pid / (510 * 510);

    const float* base = in + ((size_t)(n * 3) * 512 + y) * 512 + x;
    float r[27];
#pragma unroll
    for (int ic = 0; ic < 3; ++ic)
#pragma unroll
        for (int ky = 0; ky < 3; ++ky)
#pragma unroll
            for (int kx = 0; kx < 3; ++kx)
                r[ic * 9 + ky * 3 + kx] = base[ic * 262144 + ky * 512 + kx];

    union { signed char c[16]; int4 v4; } u;
#pragma unroll
    for (int oc = 0; oc < 16; ++oc) {
        float acc = 0.0f;
#pragma unroll
        for (int j = 0; j < 27; ++j) acc += r[j] * ws[oc * 27 + j];
        u.c[oc] = fsign8(acc);
    }
    *(int4*)(s1 + (size_t)pid * 16) = u.v4;
}

__global__ __launch_bounds__(128) void layer2_kernel(const signed char* __restrict__ s1,
                                                     const v8i* __restrict__ bp2,
                                                     signed char* __restrict__ s2) {
    __shared__ int tilebuf_i[4][192];
    int lane = threadIdx.x & 31;
    int wv   = threadIdx.x >> 5;
    int tile = blockIdx.x * 4 + wv;
    int x0 = (tile & 15) * 32;
    int t2 = tile >> 4;
    int y = t2 % 508;
    int n = t2 / 508;
    int m = lane & 15;
    int h = lane >> 4;
    int xe0 = x0 + m;      if (xe0 > 507) xe0 = 507;
    int xe1 = x0 + 16 + m; if (xe1 > 507) xe1 = 507;
    int rowbase = (n * 510 + y) * 510;
    const signed char* ab0 = s1 + (size_t)(rowbase + xe0) * 16 + h * 8;
    const signed char* ab1 = s1 + (size_t)(rowbase + xe1) * 16 + h * 8;

    v8i acc00 = {0, 0, 0, 0, 0, 0, 0, 0};
    v8i acc10 = acc00, acc01 = acc00, acc11 = acc00;
#pragma unroll
    for (int s = 0; s < 3; ++s) {
        v8i a0, a1;
#pragma unroll
        for (int v = 0; v < 8; ++v) {
            int K0c = s * 64 + (v >> 1) * 16 + (v & 1) * 4;
            int w0 = 0, w1 = 0;
            if (K0c < 144) {
                int p = K0c >> 4;
                int ky = p / 3, kx = p % 3;
                int ofs = (ky * 510 + kx) * 16 + (v & 1) * 4;
                w0 = *(const int*)(ab0 + ofs);
                w1 = *(const int*)(ab1 + ofs);
            }
            a0[v] = w0; a1[v] = w1;
        }
        v8i b0 = bp2[(0 * 3 + s) * 32 + lane];
        v8i b1 = bp2[(1 * 3 + s) * 32 + lane];
        acc00 = __builtin_amdgcn_wmma_i32_16x16x64_iu8(true, a0, true, b0, acc00, false, false);
        acc10 = __builtin_amdgcn_wmma_i32_16x16x64_iu8(true, a1, true, b0, acc10, false, false);
        acc01 = __builtin_amdgcn_wmma_i32_16x16x64_iu8(true, a0, true, b1, acc01, false, false);
        acc11 = __builtin_amdgcn_wmma_i32_16x16x64_iu8(true, a1, true, b1, acc11, false, false);
    }

    signed char* tbc = (signed char*)tilebuf_i[wv];
#pragma unroll
    for (int v = 0; v < 8; ++v) {
        int pix0 = v + 8 * h;
        int pix1 = 16 + pix0;
        tbc[pix0 * 24 + m] = clamp8(acc00[v]);
        tbc[pix1 * 24 + m] = clamp8(acc10[v]);
        signed char q0 = (m < 7) ? clamp8(acc01[v]) : (signed char)0;
        signed char q1 = (m < 7) ? clamp8(acc11[v]) : (signed char)0;
        if (m < 8) {
            tbc[pix0 * 24 + 16 + m] = q0;
            tbc[pix1 * 24 + 16 + m] = q1;
        }
    }
    asm volatile("s_wait_dscnt 0x0" ::: "memory");

    int valid = 508 - x0; if (valid > 32) valid = 32;
    int nd = valid * 6;
    const int* tb = tilebuf_i[wv];
    signed char* dst = s2 + (size_t)((n * 508 + y) * 508 + x0) * 24;
#pragma unroll
    for (int i = 0; i < 6; ++i) {
        int dw = lane + i * 32;
        if (dw < nd) *(int*)(dst + dw * 4) = tb[dw];
    }
}

__global__ __launch_bounds__(128) void layer3_kernel(const signed char* __restrict__ s2,
                                                     const v8i* __restrict__ bp3,
                                                     float* __restrict__ out) {
    __shared__ float obuf[4][64];
    int lane = threadIdx.x & 31;
    int wv   = threadIdx.x >> 5;
    int tile = blockIdx.x * 4 + wv;
    int x0 = (tile & 15) * 32;
    int t2 = tile >> 4;
    int y = t2 % 506;
    int n = t2 / 506;
    int m = lane & 15;
    int h = lane >> 4;
    int xe0 = x0 + m;      if (xe0 > 505) xe0 = 505;
    int xe1 = x0 + 16 + m; if (xe1 > 505) xe1 = 505;
    int rowbase = (n * 508 + y) * 508;
    const signed char* ab0 = s2 + (size_t)(rowbase + xe0) * 24 + h * 4;
    const signed char* ab1 = s2 + (size_t)(rowbase + xe1) * 24 + h * 4;

    v8i acc0 = {0, 0, 0, 0, 0, 0, 0, 0};
    v8i acc1 = acc0;
#pragma unroll
    for (int s = 0; s < 4; ++s) {
        v8i a0, a1;
#pragma unroll
        for (int v = 0; v < 8; ++v) {
            int K0c = s * 64 + (v >> 1) * 16 + (v & 1) * 4;
            int dc = K0c >> 2;
            int q = ((dc >> 2) << 1) + (dc & 1);
            int w0 = 0, w1 = 0;
            if (q <= 26) {
                int p = q / 3;
                int ky = p / 3, kx = p % 3;
                int ofs = (ky * 508 + kx) * 24 + (q % 3) * 8;
                w0 = *(const int*)(ab0 + ofs);
                w1 = *(const int*)(ab1 + ofs);
            }
            a0[v] = w0; a1[v] = w1;
        }
        v8i b = bp3[s * 32 + lane];
        acc0 = __builtin_amdgcn_wmma_i32_16x16x64_iu8(true, a0, true, b, acc0, false, false);
        acc1 = __builtin_amdgcn_wmma_i32_16x16x64_iu8(true, a1, true, b, acc1, false, false);
    }

    if (m < 2) {
#pragma unroll
        for (int v = 0; v < 8; ++v) {
            int q0 = acc0[v]; q0 = q0 > 1 ? 1 : (q0 < -1 ? -1 : q0);
            int q1 = acc1[v]; q1 = q1 > 1 ? 1 : (q1 < -1 ? -1 : q1);
            obuf[wv][m * 32 + v + 8 * h]      = (float)q0;
            obuf[wv][m * 32 + 16 + v + 8 * h] = (float)q1;
        }
    }
    asm volatile("s_wait_dscnt 0x0" ::: "memory");

#pragma unroll
    for (int i = 0; i < 2; ++i) {
        int dw = lane + i * 32;
        int oc = dw >> 5, px = dw & 31;
        int xx = x0 + px;
        if (xx < 506)
            out[((n * 2 + oc) * 506 + y) * 506 + xx] = obuf[wv][dw];
    }
}

extern "C" void kernel_launch(void* const* d_in, const int* in_sizes, int n_in,
                              void* d_out, int out_size, void* d_ws, size_t ws_size,
                              hipStream_t stream) {
    const float* in = (const float*)d_in[0];
    const float* w1 = (const float*)d_in[1];
    const float* w2 = (const float*)d_in[2];
    const float* w3 = (const float*)d_in[3];
    float* out = (float*)d_out;

    char* ws = (char*)d_ws;
    signed char* s1  = (signed char*)(ws);
    signed char* s2  = (signed char*)(ws + 133171200L);
    unsigned int* bp2 = (unsigned int*)(ws + 331364352L);
    unsigned int* bp3 = (unsigned int*)(ws + 331370496L);

    bpack2_kernel<<<6, 256, 0, stream>>>(w2, bp2);
    bpack3_kernel<<<4, 256, 0, stream>>>(w3, bp3);

    layer1_kernel<<<32513, 256, 0, stream>>>(in, w1, s1);

    layer2_kernel<<<65024, 128, 0, stream>>>(s1, (const v8i*)bp2, s2);

    layer3_kernel<<<64768, 128, 0, stream>>>(s2, (const v8i*)bp3, out);
}
